// QuantConvCH_28810640621898
// MI455X (gfx1250) — 8-bit matrix operand order repaired and checked by reading, compiled, not run on hardware
//
#include <hip/hip_runtime.h>
#include <hip/hip_bf16.h>

typedef int v8i __attribute__((ext_vector_type(8)));

#define NB   32
#define NCH  256
#define HH   56
#define WW   56
#define PW   58
#define NG   16
#define PG   16
#define QMAXF 127.0f
#define PCOLS 34

__global__ __launch_bounds__(256) void k_minmax(const float* __restrict__ x,
                                                float* __restrict__ xmin,
                                                float* __restrict__ xmax) {
    int c = blockIdx.x, tid = threadIdx.x;
    float mn = __FLT_MAX__, mx = -__FLT_MAX__;
    for (int b = 0; b < NB; ++b) {
        const float* p = x + ((size_t)b * NCH + c) * (HH * WW);
        for (int i = tid; i < HH * WW; i += 256) {
            float v = p[i];
            mn = fminf(mn, v); mx = fmaxf(mx, v);
        }
    }
    __shared__ float smn[256], smx[256];
    smn[tid] = mn; smx[tid] = mx; __syncthreads();
    for (int s = 128; s > 0; s >>= 1) {
        if (tid < s) { smn[tid] = fminf(smn[tid], smn[tid + s]);
                       smx[tid] = fmaxf(smx[tid], smx[tid + s]); }
        __syncthreads();
    }
    if (tid == 0) { xmin[c] = smn[0]; xmax[c] = smx[0]; }
}

__global__ __launch_bounds__(256) void k_prep(const float* __restrict__ xmin,
                                              const float* __restrict__ xmax,
                                              int*   __restrict__ perm,
                                              float* __restrict__ meanx,
                                              float* __restrict__ ascale,
                                              float* __restrict__ qsub,
                                              float* __restrict__ qinv) {
    __shared__ float key[256]; __shared__ int pidx[256];
    __shared__ float hr[256];  __shared__ float ascs[16];
    int c = threadIdx.x;
    float mn = xmin[c], mx = xmax[c];
    meanx[c] = 0.5f * (mn + mx);
    key[c] = fabsf(mx - mn);
    __syncthreads();
    float kc = key[c]; int rank = 0;
    for (int j = 0; j < 256; ++j) {
        float kj = key[j];
        rank += (kj < kc) || (kj == kc && j < c);
    }
    pidx[rank] = c;
    __syncthreads();
    int sc = pidx[c];
    perm[c] = sc;
    hr[c]   = 0.5f * (xmax[sc] - xmin[sc]);
    qsub[c] = 0.5f * (xmax[sc] + xmin[sc]);
    __syncthreads();
    if (c < NG) {
        float mg = hr[c * PG];
        for (int i = 1; i < PG; ++i) mg = fmaxf(mg, hr[c * PG + i]);
        float as = mg / QMAXF;
        ascs[c] = as; ascale[c] = as;
    }
    __syncthreads();
    float as = ascs[c >> 4];
    qinv[c] = (as > 0.f) ? (1.0f / as) : 0.f;
}

__global__ __launch_bounds__(256) void k_resid(const float* __restrict__ wq,
                                               const float* __restrict__ meanx,
                                               const float* __restrict__ bias,
                                               const float* __restrict__ wscale_p,
                                               float* __restrict__ rtab) {
    int o = blockIdx.x, c = threadIdx.x;
    float m = meanx[c];
    const float* wp = wq + ((size_t)o * NCH + c) * 9;
    float s[9];
#pragma unroll
    for (int t = 0; t < 9; ++t) s[t] = m * wp[t];
    __shared__ float red[256]; __shared__ float tapsum[9];
    for (int t = 0; t < 9; ++t) {
        red[c] = s[t]; __syncthreads();
        for (int st = 128; st > 0; st >>= 1) {
            if (c < st) red[c] += red[c + st];
            __syncthreads();
        }
        if (c == 0) tapsum[t] = red[0];
        __syncthreads();
    }
    if (c < 9) {
        int ry = c / 3, rx = c % 3;
        float acc = 0.f;
#pragma unroll
        for (int kh = 0; kh < 3; ++kh) {
            bool khok = (ry == 0) ? (kh >= 1) : ((ry == 2) ? (kh <= 1) : true);
#pragma unroll
            for (int kw = 0; kw < 3; ++kw) {
                bool kwok = (rx == 0) ? (kw >= 1) : ((rx == 2) ? (kw <= 1) : true);
                if (khok && kwok) acc += tapsum[kh * 3 + kw];
            }
        }
        rtab[o * 9 + c] = bias[o] + (*wscale_p) * acc;
    }
}

__global__ __launch_bounds__(256) void k_zero(uint4* __restrict__ p, int n4) {
    int i = blockIdx.x * 256 + threadIdx.x;
    if (i < n4) { uint4 z; z.x = z.y = z.z = z.w = 0u; p[i] = z; }
}

__global__ __launch_bounds__(256) void k_quant(const float* __restrict__ x,
                                               const int*   __restrict__ perm,
                                               const float* __restrict__ qsub,
                                               const float* __restrict__ qinv,
                                               char* __restrict__ xqp) {
    int blk = blockIdx.x;
    int cp = blk & 255, b = blk >> 8;
    int sc = perm[cp];
    float sub = qsub[cp], inv = qinv[cp];
    const float* p = x + ((size_t)b * NCH + sc) * (HH * WW);
    for (int i = threadIdx.x; i < HH * WW; i += 256) {
        int y = i / WW, xx = i - y * WW;
        int q = __float2int_rn((p[i] - sub) * inv);
        q = max(-128, min(127, q));
        xqp[(((size_t)b * PW + (y + 1)) * PW + (xx + 1)) * NCH + cp] = (char)q;
    }
}

__global__ __launch_bounds__(256) void k_packw(const float* __restrict__ wq,
                                               const int*   __restrict__ perm,
                                               int* __restrict__ wpk) {
    int idx = blockIdx.x * 256 + threadIdx.x;
    int d = idx & 7;       int rest = idx >> 3;
    int lane = rest & 31;  rest >>= 5;
    int ck = rest % 3;     rest /= 3;
    int g = rest & 15;     int ot = rest >> 4;
    int n = ot * 16 + (lane & 15);
    int Kbase = ((d >> 1) << 4) + ((lane >> 4) << 3) + ((d & 1) << 2);
    unsigned wrd = 0;
#pragma unroll
    for (int bb = 0; bb < 4; ++bb) {
        int kk = ck * 64 + Kbase + bb;
        int vq = 0;
        if (kk < 144) {
            int tap = kk >> 4, ch = kk & 15;
            int cp = g * PG + ch;
            int src = perm[cp];
            int kh = tap / 3, kw = tap - 3 * kh;
            int v = __float2int_rn(wq[(((size_t)n * NCH + src) * 3 + kh) * 3 + kw]);
            vq = max(-128, min(127, v));
        }
        wrd |= ((unsigned)(vq & 0xff)) << (bb * 8);
    }
    wpk[idx] = (int)wrd;
}

__global__ __launch_bounds__(128) void k_conv(const char* __restrict__ xqp,
                                              const char* __restrict__ wpkb,
                                              const float* __restrict__ ascale,
                                              const float* __restrict__ rtab,
                                              const float* __restrict__ wscale_p,
                                              float* __restrict__ out) {
    __shared__ __attribute__((aligned(16))) char patch[3 * PCOLS * NCH];
    __shared__ float asc[NG];
    int tid = threadIdx.x;
    int blk = blockIdx.x;
    int xt = blk & 1;
    int y  = (blk >> 1) % HH;
    int b  = blk / (HH * 2);
    int x0 = xt * 32;

    if (tid < NG) asc[tid] = ascale[tid];

    size_t gbase = ((size_t)b * PW + y) * PW * NCH;
    for (int i = tid; i < 3 * PCOLS * 16; i += 128) {
        int r = i / (PCOLS * 16);
        int rem = i - r * (PCOLS * 16);
        int col = rem >> 4;
        int c16 = (rem & 15) << 4;
        int px = x0 + col;
        uint4 v; v.x = v.y = v.z = v.w = 0u;
        if (px < PW)
            v = *(const uint4*)(xqp + gbase + (size_t)r * (PW * NCH) +
                                (size_t)px * NCH + c16);
        *(uint4*)(patch + (r * PCOLS + col) * NCH + c16) = v;
    }
    __syncthreads();

    int lane = tid & 31;
    int wave = tid >> 5;
    int m    = lane & 15;
    int hiL  = lane >> 4;

    const char* pbase = patch + m * NCH + hiL * 8;
    const char* wb = wpkb + ((size_t)wave * 4 * 16 * 3 * 32 + lane) * 32;

    float facc[2][4][8] = {};

    v8i A8a = {}; v8i A8b = {};
    v8i B8[4];
#pragma unroll
    for (int t = 0; t < 4; ++t) B8[t] = (v8i){};

    for (int g = 0; g < NG; ++g) {
        const char* wbg = wb + g * 3072;
        const char* pg  = pbase + g * PG;
#pragma unroll
        for (int t = 0; t < 4; ++t)
            __builtin_prefetch(wbg + t * 49152 + 3072, 0, 0);

        v8i ia[2][4];
#pragma unroll
        for (int mt = 0; mt < 2; ++mt)
#pragma unroll
            for (int t = 0; t < 4; ++t) ia[mt][t] = (v8i){};

#pragma unroll
        for (int ck = 0; ck < 2; ++ck) {
            v8i A0, A1;
#pragma unroll
            for (int q = 0; q < 4; ++q) {
                int tap = ck * 4 + q;
                int kh = tap / 3, kw = tap - 3 * kh;
                int off = (kh * PCOLS + kw) * NCH;
                const int2 h0 = *(const int2*)(pg + off);
                const int2 h1 = *(const int2*)(pg + off + 16 * NCH);
                A0[2 * q] = h0.x; A0[2 * q + 1] = h0.y;
                A1[2 * q] = h1.x; A1[2 * q + 1] = h1.y;
            }
            v8i Bv[4];
#pragma unroll
            for (int t = 0; t < 4; ++t) {
                int4 blo = *(const int4*)(wbg + t * 49152 + ck * 1024);
                int4 bhi = *(const int4*)(wbg + t * 49152 + ck * 1024 + 16);
                Bv[t][0]=blo.x; Bv[t][1]=blo.y; Bv[t][2]=blo.z; Bv[t][3]=blo.w;
                Bv[t][4]=bhi.x; Bv[t][5]=bhi.y; Bv[t][6]=bhi.z; Bv[t][7]=bhi.w;
            }
#pragma unroll
            for (int t = 0; t < 4; ++t) {
                ia[0][t] = __builtin_amdgcn_wmma_i32_16x16x64_iu8(true, A0, true, Bv[t], ia[0][t], false, false);
                ia[1][t] = __builtin_amdgcn_wmma_i32_16x16x64_iu8(true, A1, true, Bv[t], ia[1][t], false, false);
            }
        }
        {
            const int off = (2 * PCOLS + 2) * NCH;
            const int2 h0 = *(const int2*)(pg + off);
            const int2 h1 = *(const int2*)(pg + off + 16 * NCH);
            A8a[0] = h0.x; A8a[1] = h0.y;
            A8b[0] = h1.x; A8b[1] = h1.y;
#pragma unroll
            for (int t = 0; t < 4; ++t) {
                int4 c0 = *(const int4*)(wbg + t * 49152 + 2048);
                B8[t][0]=c0.x; B8[t][1]=c0.y; B8[t][2]=c0.z; B8[t][3]=c0.w;
            }
#pragma unroll
            for (int t = 0; t < 4; ++t) {
                ia[0][t] = __builtin_amdgcn_wmma_i32_16x16x64_iu8(true, A8a, true, B8[t], ia[0][t], false, false);
                ia[1][t] = __builtin_amdgcn_wmma_i32_16x16x64_iu8(true, A8b, true, B8[t], ia[1][t], false, false);
            }
        }
        float s = asc[g];
#pragma unroll
        for (int mt = 0; mt < 2; ++mt)
#pragma unroll
            for (int t = 0; t < 4; ++t)
#pragma unroll
                for (int j = 0; j < 8; ++j)
                    facc[mt][t][j] += s * (float)ia[mt][t][j];
    }

    float wsv = *wscale_p;
    int ry = (y == 0) ? 0 : ((y == HH - 1) ? 2 : 1);
#pragma unroll
    for (int t = 0; t < 4; ++t) {
        int o = (wave * 4 + t) * 16 + (lane & 15);
        const float* rb = rtab + o * 9 + ry * 3;
        float* ob = out + (((size_t)b * NCH + o) * HH + y) * WW;
#pragma unroll
        for (int mt = 0; mt < 2; ++mt) {
#pragma unroll
            for (int j = 0; j < 8; ++j) {
                int px = x0 + mt * 16 + j + hiL * 8;
                if (px < WW) {
                    int rx = (px == 0) ? 0 : ((px == WW - 1) ? 2 : 1);
                    ob[px] = facc[mt][t][j] * wsv + rb[rx];
                }
            }
        }
    }
}

extern "C" void kernel_launch(void* const* d_in, const int* in_sizes, int n_in,
                              void* d_out, int out_size, void* d_ws, size_t ws_size,
                              hipStream_t stream) {
    const float* x      = (const float*)d_in[0];
    const float* wq     = (const float*)d_in[1];
    const float* wscale = (const float*)d_in[2];
    const float* bias   = (const float*)d_in[3];
    float* out = (float*)d_out;
    char* ws = (char*)d_ws;

    float* xmin   = (float*)(ws + 0);
    float* xmax   = (float*)(ws + 1024);
    float* meanx  = (float*)(ws + 2048);
    int*   perm   = (int*)  (ws + 3072);
    float* ascale = (float*)(ws + 4096);
    float* qsub   = (float*)(ws + 4352);
    float* qinv   = (float*)(ws + 5376);
    float* rtab   = (float*)(ws + 6400);
    int*   wpk    = (int*)  (ws + 16384);
    char*  xqp    = ws + 802816;

    k_minmax<<<256, 256, 0, stream>>>(x, xmin, xmax);
    k_prep  <<<1,   256, 0, stream>>>(xmin, xmax, perm, meanx, ascale, qsub, qinv);
    k_resid <<<256, 256, 0, stream>>>(wq, meanx, bias, wscale, rtab);

    int n4 = (NB * PW * PW * NCH) / 16;
    k_zero  <<<(n4 + 255) / 256, 256, 0, stream>>>((uint4*)xqp, n4);
    k_quant <<<NB * NCH, 256, 0, stream>>>(x, perm, qsub, qinv, xqp);
    k_packw <<<768, 256, 0, stream>>>(wq, perm, wpk);

    k_conv  <<<NB * HH * 2, 128, 0, stream>>>(xqp, (const char*)wpk,
                                              ascale, rtab, wscale, out);
}
